// DeepGATGNN_47974784696415
// MI455X (gfx1250) — matrix operand order repaired and checked by reading, not compiled, not run on hardware
//
#include <hip/hip_runtime.h>
#include <math.h>

#define N_NODES 50000
#define N_EDGES 300000
#define N_GRAPHS 500
#define F_NODE 92
#define F_EDGE 50
#define F_GLOBAL 108
#define HID 64
#define HEADS 4
#define HU 256
#define LAYERS 5
#define GIN (HID + F_GLOBAL)   // 172

typedef __attribute__((ext_vector_type(16))) __bf16 v16bf;
typedef __attribute__((ext_vector_type(8)))  float  v8f;
typedef unsigned int u32x4 __attribute__((ext_vector_type(4)));
typedef int          i32x4 __attribute__((ext_vector_type(4)));
typedef int          i32x8 __attribute__((ext_vector_type(8)));

__device__ __forceinline__ unsigned short f2bf(float f) {
  unsigned int u = __float_as_uint(f);
  unsigned int r = (u + 0x7FFFu + ((u >> 16) & 1u)) >> 16;   // RNE
  return (unsigned short)r;
}

// ---------------------------------------------------------------------------
// TDM: async 2D bf16 tile load (tile = 64 rows x 32 cols) global -> LDS.
// D# built per CDNA5 ISA 8.3-8.6. OOB rows/cols (beyond tensor_w/tensor_h)
// are zero-filled by hardware == our tail padding. pad_interval=16 DWORDs,
// pad_amount=1 DWORD reproduces the [64][34]-ushort LDS stride.
// Tracked by TENSORcnt (one wave issues; s_wait_tensorcnt + barrier syncs).
// This toolchain's builtin takes 6 args (extra i32x8 group, zero for 2D).
// ---------------------------------------------------------------------------
__device__ __forceinline__ void tdm_load_tile_bf16(const void* lds_dst,
                                                   const unsigned short* gsrc,
                                                   int tensor_w, int tensor_h,
                                                   long long row_stride_elems)
{
  unsigned long long ga = (unsigned long long)gsrc;
  unsigned ldsa = (unsigned)(unsigned long long)lds_dst;   // low 32 bits of flat LDS addr = offset
  unsigned td0 = (unsigned)(tensor_w > 0 ? tensor_w : 0);
  unsigned td1 = (unsigned)(tensor_h > 0 ? tensor_h : 0);
  unsigned long long st = (unsigned long long)row_stride_elems;

  u32x4 g0;
  g0[0] = 1u;                                           // count=1 (valid), user mode
  g0[1] = ldsa;                                         // lds_addr
  g0[2] = (unsigned)(ga & 0xFFFFFFFFull);               // global_addr[31:0]
  g0[3] = (unsigned)((ga >> 32) & 0x01FFFFFFull)        // global_addr[56:32]
        | (2u << 30);                                   // type=2 ("image")
  i32x8 g1;
  g1[0] = (int)((1u << 16)                              // data_size = 2 bytes
              | (1u << 20)                              // pad_enable
              | (3u << 22));                            // pad_interval: 16 DWORDs
                                                        // pad_amount field 0 -> 1 DWORD
  g1[1] = (int)((td0 & 0xFFFFu) << 16);                 // tensor_dim0[15:0]
  g1[2] = (int)((td0 >> 16) | ((td1 & 0xFFFFu) << 16)); // tensor_dim0[31:16] | tensor_dim1[15:0]
  g1[3] = (int)((td1 >> 16) | (32u << 16));             // tensor_dim1[31:16] | tile_dim0=32
  g1[4] = 64;                                           // tile_dim1=64, tile_dim2=0
  g1[5] = (int)(st & 0xFFFFFFFFull);                    // tensor_dim0_stride[31:0]
  g1[6] = (int)((st >> 32) & 0xFFFFull);                // tensor_dim0_stride[47:32]
  g1[7] = 0;
  i32x4 z4 = {0, 0, 0, 0};                              // 2D: groups 2/3 unused
  i32x8 z8 = {0, 0, 0, 0, 0, 0, 0, 0};                  // extra group (clang-23 form)
  __builtin_amdgcn_tensor_load_to_lds(g0, g1, z4, z4, z8, 0);
}

// ---------------------------------------------------------------------------
// GEMM: C[M,N] = bf16(A[M,K]) @ bf16(Bt[N,K])^T + bias, optional ReLU.
// A, Bt are bf16 in memory (halved HBM/L2 traffic). Tiles staged by TDM,
// double-buffered so DMA overlaps WMMA. Block = 128 threads (4 waves),
// tile 64x64, BK=32; wave w owns rows [16w,16w+16) as 4 WMMA accumulators.
// ---------------------------------------------------------------------------
__global__ void gemm_bf16_wmma(const unsigned short* __restrict__ A,
                               const unsigned short* __restrict__ Bt,
                               const float* __restrict__ bias,
                               float* __restrict__ C,
                               unsigned short* __restrict__ Cbf,
                               int M, int N, int K, int relu)
{
  __shared__ unsigned short As[2][64][34];
  __shared__ unsigned short Bs[2][64][34];
  const int m0   = blockIdx.x * 64;
  const int n0   = blockIdx.y * 64;
  const int tid  = threadIdx.x;
  const int wave = tid >> 5;
  const int lane = tid & 31;
  const int ntiles = (K + 31) >> 5;

  v8f acc[4];
  #pragma unroll
  for (int t = 0; t < 4; ++t)
    #pragma unroll
    for (int j = 0; j < 8; ++j) acc[t][j] = 0.0f;

  if (wave == 0) {
    tdm_load_tile_bf16(&As[0][0][0], A + (size_t)m0 * K, K, M - m0, K);
    tdm_load_tile_bf16(&Bs[0][0][0], Bt + (size_t)n0 * K, K, N - n0, K);
  }
  for (int kt = 0; kt < ntiles; ++kt) {
    const int buf = kt & 1;
    if (wave == 0) {
      if (kt + 1 < ntiles) {
        const int k1 = (kt + 1) << 5;
        tdm_load_tile_bf16(&As[buf ^ 1][0][0], A + (size_t)m0 * K + k1, K - k1, M - m0, K);
        tdm_load_tile_bf16(&Bs[buf ^ 1][0][0], Bt + (size_t)n0 * K + k1, K - k1, N - n0, K);
        __builtin_amdgcn_s_wait_tensorcnt((short)2);   // current buf's 2 loads done
      } else {
        __builtin_amdgcn_s_wait_tensorcnt((short)0);
      }
    }
    __syncthreads();

    const int am   = (wave << 4) + (lane & 15);
    const int koff = (lane >> 4) << 3;
    union { unsigned int u[8]; v16bf v; } af;
    #pragma unroll
    for (int j = 0; j < 4; ++j) {
      af.u[j]     = *(const unsigned int*)&As[buf][am][koff + 2 * j];
      af.u[j + 4] = *(const unsigned int*)&As[buf][am][16 + koff + 2 * j];
    }
    #pragma unroll
    for (int t = 0; t < 4; ++t) {
      const int bn = (t << 4) + (lane & 15);
      const int bk = (lane >> 4) << 3;
      union { unsigned int u[8]; v16bf v; } bfr;
      #pragma unroll
      for (int j = 0; j < 8; ++j)
        bfr.u[j] = *(const unsigned int*)&Bs[buf][bn][((j & 4) ? 16 : 0) + bk + 2 * (j & 3)];
      acc[t] = __builtin_amdgcn_wmma_f32_16x16x32_bf16(
          false, af.v, false, bfr.v, (short)0, acc[t], false, false);
    }
    __syncthreads();
  }

  const int rbase = (wave << 4) + ((lane >> 4) << 3);
  #pragma unroll
  for (int t = 0; t < 4; ++t) {
    int gn = n0 + (t << 4) + (lane & 15);
    if (gn >= N) continue;
    float bv = bias ? bias[gn] : 0.0f;
    #pragma unroll
    for (int r = 0; r < 8; ++r) {
      int gm = m0 + rbase + r;
      if (gm < M) {
        float v = acc[t][r] + bv;
        if (relu) v = fmaxf(v, 0.0f);
        if (C)   C[(size_t)gm * N + gn] = v;
        if (Cbf) Cbf[(size_t)gm * N + gn] = f2bf(v);
      }
    }
  }
}

// ---------------------------------------------------------------------------
// Fused edge projection + attention scores (one head per blockIdx.y):
// scores[e,h] = sum_u (eh@We + be + q[src])_u * k[dst]_u over this head's 64
// cols — reduction stays in-block, never materializing the 307 MB `el`.
// ---------------------------------------------------------------------------
__global__ void edge_score_wmma(const unsigned short* __restrict__ ehbf,
                                const unsigned short* __restrict__ Wet,  // [256][64] bf16
                                const float* __restrict__ bel,
                                const float* __restrict__ q,
                                const float* __restrict__ k,
                                const int* __restrict__ esrc,
                                const int* __restrict__ edst,
                                float* __restrict__ scores)
{
  __shared__ unsigned short As[2][64][34];
  __shared__ unsigned short Bs[2][64][34];
  const int m0   = blockIdx.x * 64;
  const int h    = blockIdx.y;
  const int n0   = h * HID;
  const int tid  = threadIdx.x;
  const int wave = tid >> 5;
  const int lane = tid & 31;

  v8f acc[4];
  #pragma unroll
  for (int t = 0; t < 4; ++t)
    #pragma unroll
    for (int j = 0; j < 8; ++j) acc[t][j] = 0.0f;

  if (wave == 0) {
    tdm_load_tile_bf16(&As[0][0][0], ehbf + (size_t)m0 * HID, HID, N_EDGES - m0, HID);
    tdm_load_tile_bf16(&Bs[0][0][0], Wet + (size_t)n0 * HID, HID, HU - n0, HID);
  }
  #pragma unroll
  for (int kt = 0; kt < 2; ++kt) {   // K = 64
    const int buf = kt & 1;
    if (wave == 0) {
      if (kt == 0) {
        tdm_load_tile_bf16(&As[1][0][0], ehbf + (size_t)m0 * HID + 32, HID - 32, N_EDGES - m0, HID);
        tdm_load_tile_bf16(&Bs[1][0][0], Wet + (size_t)n0 * HID + 32, HID - 32, HU - n0, HID);
        __builtin_amdgcn_s_wait_tensorcnt((short)2);
      } else {
        __builtin_amdgcn_s_wait_tensorcnt((short)0);
      }
    }
    __syncthreads();
    const int am   = (wave << 4) + (lane & 15);
    const int koff = (lane >> 4) << 3;
    union { unsigned int u[8]; v16bf v; } af;
    #pragma unroll
    for (int j = 0; j < 4; ++j) {
      af.u[j]     = *(const unsigned int*)&As[buf][am][koff + 2 * j];
      af.u[j + 4] = *(const unsigned int*)&As[buf][am][16 + koff + 2 * j];
    }
    #pragma unroll
    for (int t = 0; t < 4; ++t) {
      const int bn = (t << 4) + (lane & 15);
      const int bk = (lane >> 4) << 3;
      union { unsigned int u[8]; v16bf v; } bfr;
      #pragma unroll
      for (int j = 0; j < 8; ++j)
        bfr.u[j] = *(const unsigned int*)&Bs[buf][bn][((j & 4) ? 16 : 0) + bk + 2 * (j & 3)];
      acc[t] = __builtin_amdgcn_wmma_f32_16x16x32_bf16(
          false, af.v, false, bfr.v, (short)0, acc[t], false, false);
    }
    __syncthreads();
  }

  const int half  = lane >> 4;
  const int nl    = lane & 15;
  const int rbase = (wave << 4) + (half << 3);
  #pragma unroll
  for (int r = 0; r < 8; ++r) {
    int gm = m0 + rbase + r;
    float p = 0.0f;
    if (gm < N_EDGES) {
      int s = esrc[gm], d = edst[gm];
      const float* qrow = q + (size_t)s * HU + n0;
      const float* krow = k + (size_t)d * HU + n0;
      #pragma unroll
      for (int t = 0; t < 4; ++t) {
        int c = (t << 4) + nl;
        p += (acc[t][r] + bel[n0 + c] + qrow[c]) * krow[c];
      }
    }
    #pragma unroll
    for (int off = 1; off < 16; off <<= 1) p += __shfl_xor(p, off, 32);
    if (nl == 0 && gm < N_EDGES) scores[(size_t)gm * HEADS + h] = p;
  }
}

// f32 -> bf16 buffer conversion
__global__ void cvt_kernel(const float* __restrict__ in, unsigned short* __restrict__ out,
                           size_t n)
{
  size_t i = blockIdx.x * (size_t)blockDim.x + threadIdx.x;
  if (i < n) out[i] = f2bf(in[i]);
}

// transpose + convert weight: Bt[n*K + k] = bf16(B[k*N + n])
__global__ void tcvt_kernel(const float* __restrict__ B, unsigned short* __restrict__ Bt,
                            int K, int N)
{
  int i = blockIdx.x * blockDim.x + threadIdx.x;
  if (i >= K * N) return;
  int n = i / K, kk = i % K;
  Bt[i] = f2bf(B[(size_t)kk * N + n]);
}

__global__ void lower_bound_kernel(const int* __restrict__ sorted, int n_sorted,
                                   int n_vals, int* __restrict__ offs)
{
  int i = blockIdx.x * blockDim.x + threadIdx.x;
  if (i > n_vals) return;
  int lo = 0, hi = n_sorted;
  while (lo < hi) { int mid = (lo + hi) >> 1; if (sorted[mid] < i) lo = mid + 1; else hi = mid; }
  offs[i] = lo;
}

// One wave32 per (node,head): segment softmax over that node's edges + weighted
// aggregation of v[dst]; emits bf16 agg directly (only consumed by Wo GEMM).
__global__ void softmax_agg(const float* __restrict__ scores,
                            const float* __restrict__ v,
                            const int* __restrict__ edst,
                            const int* __restrict__ noff,
                            unsigned short* __restrict__ aggbf)
{
  int gwave = (int)((blockIdx.x * (size_t)blockDim.x + threadIdx.x) >> 5);
  int lane  = threadIdx.x & 31;
  if (gwave >= N_NODES * HEADS) return;
  int node = gwave >> 2;
  int h    = gwave & 3;
  int e0 = noff[node], e1 = noff[node + 1];

  float m = -INFINITY;
  for (int e = e0 + lane; e < e1; e += 32) m = fmaxf(m, scores[(size_t)e * HEADS + h]);
  #pragma unroll
  for (int off = 16; off; off >>= 1) m = fmaxf(m, __shfl_xor(m, off, 32));
  float s = 0.0f;
  for (int e = e0 + lane; e < e1; e += 32) s += __expf(scores[(size_t)e * HEADS + h] - m);
  #pragma unroll
  for (int off = 16; off; off >>= 1) s += __shfl_xor(s, off, 32);
  float inv = (s > 0.0f) ? 1.0f / s : 0.0f;

  float a0 = 0.0f, a1 = 0.0f;
  for (int e = e0; e < e1; ++e) {
    float w = __expf(scores[(size_t)e * HEADS + h] - m) * inv;
    const float* vrow = v + (size_t)edst[e] * HU + h * HID;
    a0 += w * vrow[lane];
    a1 += w * vrow[lane + 32];
  }
  unsigned short* arow = aggbf + (size_t)node * HU + h * HID;
  arow[lane]      = f2bf(a0);
  arow[lane + 32] = f2bf(a1);
}

// x_new = out + residual; residual = x_new; x = x_new*(gamma*BN_INV)+beta (f32 + bf16)
__global__ void resbn_kernel(const float* __restrict__ outb,
                             float* __restrict__ residual,
                             float* __restrict__ x,
                             unsigned short* __restrict__ xbf,
                             const float* __restrict__ gamma,
                             const float* __restrict__ beta,
                             float bninv, int n)
{
  int i = blockIdx.x * blockDim.x + threadIdx.x;
  if (i >= n) return;
  int c = i & (HID - 1);
  float t = outb[i] + residual[i];
  residual[i] = t;
  float y = t * (gamma[c] * bninv) + beta[c];
  x[i] = y;
  xbf[i] = f2bf(y);
}

// g = concat(x, global_features[graph_indices]) -> bf16 [N,172]
__global__ void concat_kernel(const float* __restrict__ x,
                              const float* __restrict__ gf,
                              const int* __restrict__ gidx,
                              unsigned short* __restrict__ g)
{
  size_t i = blockIdx.x * (size_t)blockDim.x + threadIdx.x;
  if (i >= (size_t)N_NODES * GIN) return;
  int n = (int)(i / GIN), c = (int)(i % GIN);
  float v = (c < HID) ? x[(size_t)n * HID + c]
                      : gf[(size_t)gidx[n] * F_GLOBAL + (c - HID)];
  g[i] = f2bf(v);
}

// GroupNorm(8x8) + ReLU in place; optional bf16 copy for a following GEMM.
__global__ void gnrelu_kernel(float* __restrict__ h,
                              const float* __restrict__ gamma,
                              const float* __restrict__ beta,
                              unsigned short* __restrict__ hbf, int n_rows)
{
  int i = blockIdx.x * blockDim.x + threadIdx.x;
  if (i >= n_rows * 8) return;
  int row = i >> 3, grp = i & 7;
  float* p = h + (size_t)row * HID + grp * 8;
  float vals[8], mu = 0.0f;
  #pragma unroll
  for (int j = 0; j < 8; ++j) { vals[j] = p[j]; mu += vals[j]; }
  mu *= 0.125f;
  float var = 0.0f;
  #pragma unroll
  for (int j = 0; j < 8; ++j) { float d = vals[j] - mu; var += d * d; }
  var *= 0.125f;
  float inv = rsqrtf(var + 1e-3f);
  #pragma unroll
  for (int j = 0; j < 8; ++j) {
    int c = grp * 8 + j;
    float y = fmaxf((vals[j] - mu) * inv * gamma[c] + beta[c], 0.0f);
    p[j] = y;
    if (hbf) hbf[(size_t)row * HID + c] = f2bf(y);
  }
}

__global__ void logit_kernel(const float* __restrict__ h,
                             const float* __restrict__ w,
                             const float* __restrict__ b,
                             float* __restrict__ logits, int n)
{
  int i = blockIdx.x * blockDim.x + threadIdx.x;
  if (i >= n) return;
  float s = b[0];
  const float* row = h + (size_t)i * HID;
  #pragma unroll
  for (int c = 0; c < HID; ++c) s += row[c] * w[c];
  logits[i] = s;
}

// Per-graph softmax over node logits + attention-weighted mean pool (bf16 out).
__global__ void graph_pool(const float* __restrict__ logits,
                           const float* __restrict__ x,
                           const int* __restrict__ goff,
                           unsigned short* __restrict__ pooledbf)
{
  int g    = (int)((blockIdx.x * (size_t)blockDim.x + threadIdx.x) >> 5);
  int lane = threadIdx.x & 31;
  if (g >= N_GRAPHS) return;
  int n0 = goff[g], n1 = goff[g + 1];

  float m = -INFINITY;
  for (int n = n0 + lane; n < n1; n += 32) m = fmaxf(m, logits[n]);
  #pragma unroll
  for (int off = 16; off; off >>= 1) m = fmaxf(m, __shfl_xor(m, off, 32));
  float s = 0.0f;
  for (int n = n0 + lane; n < n1; n += 32) s += __expf(logits[n] - m);
  #pragma unroll
  for (int off = 16; off; off >>= 1) s += __shfl_xor(s, off, 32);
  float cnt   = (float)(n1 - n0);
  float scale = (s > 0.0f && cnt > 0.0f) ? 1.0f / (s * cnt) : 0.0f;

  float a0 = 0.0f, a1 = 0.0f;
  for (int n = n0; n < n1; ++n) {
    float w = __expf(logits[n] - m);
    a0 += w * x[(size_t)n * HID + lane];
    a1 += w * x[(size_t)n * HID + lane + 32];
  }
  pooledbf[(size_t)g * HID + lane]      = f2bf(a0 * scale);
  pooledbf[(size_t)g * HID + lane + 32] = f2bf(a1 * scale);
}

__global__ void final_head(const float* __restrict__ p1,
                           const float* __restrict__ w,
                           const float* __restrict__ b,
                           float* __restrict__ out)
{
  int g = blockIdx.x * blockDim.x + threadIdx.x;
  if (g >= N_GRAPHS) return;
  float s = b[0];
  const float* row = p1 + (size_t)g * HID;
  #pragma unroll
  for (int c = 0; c < HID; ++c) s += row[c] * w[c];
  out[g] = s;
}

// ---------------------------------------------------------------------------
extern "C" void kernel_launch(void* const* d_in, const int* in_sizes, int n_in,
                              void* d_out, int out_size, void* d_ws, size_t ws_size,
                              hipStream_t stream) {
  const float* nf   = (const float*)d_in[0];
  const float* ef   = (const float*)d_in[1];
  const float* gf   = (const float*)d_in[2];
  const int*   esrc = (const int*)d_in[3];
  const int*   edst = (const int*)d_in[4];
  const int*   gidx = (const int*)d_in[5];
  const float* W_ne = (const float*)d_in[6];  const float* b_ne = (const float*)d_in[7];
  const float* W_ee = (const float*)d_in[8];  const float* b_ee = (const float*)d_in[9];
  const float* Wq   = (const float*)d_in[10]; const float* bq   = (const float*)d_in[11];
  const float* Wk   = (const float*)d_in[12]; const float* bk   = (const float*)d_in[13];
  const float* Wv   = (const float*)d_in[14]; const float* bv   = (const float*)d_in[15];
  const float* We   = (const float*)d_in[16]; const float* be   = (const float*)d_in[17];
  const float* Wo   = (const float*)d_in[18]; const float* bo   = (const float*)d_in[19];
  const float* bng  = (const float*)d_in[20]; const float* bnb  = (const float*)d_in[21];
  const float* gW1  = (const float*)d_in[22]; const float* gb1  = (const float*)d_in[23];
  const float* gn1g = (const float*)d_in[24]; const float* gn1b = (const float*)d_in[25];
  const float* gW2  = (const float*)d_in[26]; const float* gb2  = (const float*)d_in[27];
  const float* gn2g = (const float*)d_in[28]; const float* gn2b = (const float*)d_in[29];
  const float* gW3  = (const float*)d_in[30]; const float* gb3  = (const float*)d_in[31];
  const float* oW1  = (const float*)d_in[32]; const float* ob1  = (const float*)d_in[33];
  const float* oW2  = (const float*)d_in[34]; const float* ob2  = (const float*)d_in[35];

  char* ws = (char*)d_ws;
  size_t off = 0;
  auto alloc = [&](size_t bytes) -> void* {
    void* p = ws + off;
    off += (bytes + 255) & ~(size_t)255;
    return p;
  };
  typedef unsigned short u16;
  // f32 buffers
  float* x      = (float*)alloc((size_t)N_NODES * HID * 4);
  float* res    = (float*)alloc((size_t)N_NODES * HID * 4);
  float* outb   = (float*)alloc((size_t)N_NODES * HID * 4);
  float* qb     = (float*)alloc((size_t)N_NODES * HU * 4);
  float* kb     = (float*)alloc((size_t)N_NODES * HU * 4);
  float* vb     = (float*)alloc((size_t)N_NODES * HU * 4);
  float* scores = (float*)alloc((size_t)N_EDGES * HEADS * 4);
  float* h1     = (float*)alloc((size_t)N_NODES * HID * 4);
  float* h2     = (float*)alloc((size_t)N_NODES * HID * 4);
  float* logits = (float*)alloc((size_t)N_NODES * 4);
  float* p1     = (float*)alloc((size_t)N_GRAPHS * HID * 4);
  int*   noff   = (int*)alloc((size_t)(N_NODES + 1) * 4);
  int*   goff   = (int*)alloc((size_t)(N_GRAPHS + 1) * 4);
  // bf16 activations
  u16* nf_bf   = (u16*)alloc((size_t)N_NODES * F_NODE * 2);
  u16* ef_bf   = (u16*)alloc((size_t)N_EDGES * F_EDGE * 2);
  u16* x_bf    = (u16*)alloc((size_t)N_NODES * HID * 2);
  u16* eh_bf   = (u16*)alloc((size_t)N_EDGES * HID * 2);
  u16* agg_bf  = (u16*)alloc((size_t)N_NODES * HU * 2);
  u16* gcat_bf = (u16*)alloc((size_t)N_NODES * GIN * 2);
  u16* h1_bf   = (u16*)alloc((size_t)N_NODES * HID * 2);
  u16* pool_bf = (u16*)alloc((size_t)N_GRAPHS * HID * 2);
  // bf16 transposed weights [N][K]
  u16* Wne_t = (u16*)alloc((size_t)HID * F_NODE * 2);
  u16* Wee_t = (u16*)alloc((size_t)HID * F_EDGE * 2);
  u16* Wq_t  = (u16*)alloc((size_t)LAYERS * HU * HID * 2);
  u16* Wk_t  = (u16*)alloc((size_t)LAYERS * HU * HID * 2);
  u16* Wv_t  = (u16*)alloc((size_t)LAYERS * HU * HID * 2);
  u16* We_t  = (u16*)alloc((size_t)LAYERS * HU * HID * 2);
  u16* Wo_t  = (u16*)alloc((size_t)LAYERS * HID * HU * 2);
  u16* gW1_t = (u16*)alloc((size_t)HID * GIN * 2);
  u16* gW2_t = (u16*)alloc((size_t)HID * HID * 2);
  u16* oW1_t = (u16*)alloc((size_t)HID * HID * 2);

  const float bninv = 1.0f / sqrtf(1.0f + 1e-3f);
  const dim3 gblk(128);
  auto ggrid = [](int M, int N) { return dim3((unsigned)((M + 63) / 64), (unsigned)((N + 63) / 64)); };
  auto cvt = [&](const float* in, u16* o, size_t n) {
    cvt_kernel<<<(unsigned)((n + 255) / 256), 256, 0, stream>>>(in, o, n);
  };
  auto tcvt = [&](const float* B, u16* Bt, int K, int N) {
    tcvt_kernel<<<(K * N + 255) / 256, 256, 0, stream>>>(B, Bt, K, N);
  };

  // one-time precision/layout prep
  cvt(nf, nf_bf, (size_t)N_NODES * F_NODE);
  cvt(ef, ef_bf, (size_t)N_EDGES * F_EDGE);
  tcvt(W_ne, Wne_t, F_NODE, HID);
  tcvt(W_ee, Wee_t, F_EDGE, HID);
  for (int l = 0; l < LAYERS; ++l) {
    tcvt(Wq + (size_t)l * HID * HU, Wq_t + (size_t)l * HU * HID, HID, HU);
    tcvt(Wk + (size_t)l * HID * HU, Wk_t + (size_t)l * HU * HID, HID, HU);
    tcvt(Wv + (size_t)l * HID * HU, Wv_t + (size_t)l * HU * HID, HID, HU);
    tcvt(We + (size_t)l * HID * HU, We_t + (size_t)l * HU * HID, HID, HU);
    tcvt(Wo + (size_t)l * HU * HID, Wo_t + (size_t)l * HID * HU, HU, HID);
  }
  tcvt(gW1, gW1_t, GIN, HID);
  tcvt(gW2, gW2_t, HID, HID);
  tcvt(oW1, oW1_t, HID, HID);

  lower_bound_kernel<<<(N_NODES + 1 + 255) / 256, 256, 0, stream>>>(esrc, N_EDGES, N_NODES, noff);
  lower_bound_kernel<<<(N_GRAPHS + 1 + 255) / 256, 256, 0, stream>>>(gidx, N_NODES, N_GRAPHS, goff);

  // embeddings
  gemm_bf16_wmma<<<ggrid(N_NODES, HID), gblk, 0, stream>>>(
      nf_bf, Wne_t, b_ne, x, x_bf, N_NODES, HID, F_NODE, 0);
  gemm_bf16_wmma<<<ggrid(N_EDGES, HID), gblk, 0, stream>>>(
      ef_bf, Wee_t, b_ee, (float*)nullptr, eh_bf, N_EDGES, HID, F_EDGE, 0);
  (void)hipMemcpyAsync(res, x, (size_t)N_NODES * HID * 4, hipMemcpyDeviceToDevice, stream);

  for (int l = 0; l < LAYERS; ++l) {
    const u16* Wqt = Wq_t + (size_t)l * HU * HID; const float* bql = bq + (size_t)l * HU;
    const u16* Wkt = Wk_t + (size_t)l * HU * HID; const float* bkl = bk + (size_t)l * HU;
    const u16* Wvt = Wv_t + (size_t)l * HU * HID; const float* bvl = bv + (size_t)l * HU;
    const u16* Wet = We_t + (size_t)l * HU * HID; const float* bel = be + (size_t)l * HU;
    const u16* Wot = Wo_t + (size_t)l * HID * HU; const float* bol = bo + (size_t)l * HID;

    gemm_bf16_wmma<<<ggrid(N_NODES, HU), gblk, 0, stream>>>(
        x_bf, Wqt, bql, qb, (u16*)nullptr, N_NODES, HU, HID, 0);
    gemm_bf16_wmma<<<ggrid(N_NODES, HU), gblk, 0, stream>>>(
        x_bf, Wkt, bkl, kb, (u16*)nullptr, N_NODES, HU, HID, 0);
    gemm_bf16_wmma<<<ggrid(N_NODES, HU), gblk, 0, stream>>>(
        x_bf, Wvt, bvl, vb, (u16*)nullptr, N_NODES, HU, HID, 0);

    edge_score_wmma<<<dim3((N_EDGES + 63) / 64, HEADS), gblk, 0, stream>>>(
        eh_bf, Wet, bel, qb, kb, esrc, edst, scores);

    softmax_agg<<<(N_NODES * HEADS * 32 + 255) / 256, 256, 0, stream>>>(
        scores, vb, edst, noff, agg_bf);

    gemm_bf16_wmma<<<ggrid(N_NODES, HID), gblk, 0, stream>>>(
        agg_bf, Wot, bol, outb, (u16*)nullptr, N_NODES, HID, HU, 1);

    resbn_kernel<<<(N_NODES * HID + 255) / 256, 256, 0, stream>>>(
        outb, res, x, x_bf, bng + (size_t)l * HID, bnb + (size_t)l * HID, bninv, N_NODES * HID);
  }

  // global attention head
  concat_kernel<<<(unsigned)(((size_t)N_NODES * GIN + 255) / 256), 256, 0, stream>>>(
      x, gf, gidx, gcat_bf);
  gemm_bf16_wmma<<<ggrid(N_NODES, HID), gblk, 0, stream>>>(
      gcat_bf, gW1_t, gb1, h1, (u16*)nullptr, N_NODES, HID, GIN, 0);
  gnrelu_kernel<<<(N_NODES * 8 + 255) / 256, 256, 0, stream>>>(h1, gn1g, gn1b, h1_bf, N_NODES);
  gemm_bf16_wmma<<<ggrid(N_NODES, HID), gblk, 0, stream>>>(
      h1_bf, gW2_t, gb2, h2, (u16*)nullptr, N_NODES, HID, HID, 0);
  gnrelu_kernel<<<(N_NODES * 8 + 255) / 256, 256, 0, stream>>>(h2, gn2g, gn2b, (u16*)nullptr, N_NODES);
  logit_kernel<<<(N_NODES + 255) / 256, 256, 0, stream>>>(h2, gW3, gb3, logits, N_NODES);
  graph_pool<<<(N_GRAPHS * 32 + 255) / 256, 256, 0, stream>>>(logits, x, goff, pool_bf);

  // output MLP
  gemm_bf16_wmma<<<ggrid(N_GRAPHS, HID), gblk, 0, stream>>>(
      pool_bf, oW1_t, ob1, p1, (u16*)nullptr, N_GRAPHS, HID, HID, 1);
  final_head<<<(N_GRAPHS + 255) / 256, 256, 0, stream>>>(p1, oW2, ob2, (float*)d_out);
}
